// MPNNNet_7679401525284
// MI455X (gfx1250) — matrix operand order repaired and checked by reading, not compiled, not run on hardware
//
#include <hip/hip_runtime.h>
#include <hip/hip_bf16.h>

typedef __attribute__((ext_vector_type(16))) _Float16 v16h;
typedef __attribute__((ext_vector_type(8)))  float    v8f;
typedef __attribute__((ext_vector_type(4)))  float    v4f;

#define D 64
#define HA 32

// ---------------------------------------------------------------------------
// Fragment loaders (CDNA5 WMMA 16x16x32 f16 layouts, ISA 7.12.2)
// ---------------------------------------------------------------------------

// B fragment (32x16, f16) from f32 row-major W[64][64].
// lane n (=lane&15) holds column ntile*16+n; lanes<16: K = kb..kb+7 then kb+16..kb+23; lanes>=16: K base +8.
__device__ __forceinline__ v16h load_B_frag_f32(const float* __restrict__ W,
                                                int ntile, int kstep, int lane) {
  const int col = ntile * 16 + (lane & 15);
  const int kb  = kstep * 32 + ((lane >> 4) << 3);
  v16h b;
#pragma unroll
  for (int j = 0; j < 16; ++j)
    b[j] = (_Float16)W[(kb + (j < 8 ? j : j + 8)) * D + col];
  return b;
}

// A fragment (16x32, f16) from f32 row-major A[M][64].
// lanes<16: row=row0+lane, K = kb..kb+7 then kb+16..kb+23; lanes>=16: K base +8.
// NT=true issues non-temporal loads (single-use streams: keep them out of L2).
template <bool NT>
__device__ __forceinline__ v16h load_A_frag_f32(const float* __restrict__ A,
                                                int row0, int kstep, int lane) {
  const int m  = row0 + (lane & 15);
  const int kb = kstep * 32 + ((lane >> 4) << 3);
  const float* p = A + (size_t)m * D + kb;
  v4f q0, q1, q2, q3;
  if (NT) {
    q0 = __builtin_nontemporal_load((const v4f*)(p));
    q1 = __builtin_nontemporal_load((const v4f*)(p + 4));
    q2 = __builtin_nontemporal_load((const v4f*)(p + 16));
    q3 = __builtin_nontemporal_load((const v4f*)(p + 20));
  } else {
    q0 = *(const v4f*)(p);
    q1 = *(const v4f*)(p + 4);
    q2 = *(const v4f*)(p + 16);
    q3 = *(const v4f*)(p + 20);
  }
  v16h a;
#pragma unroll
  for (int j = 0; j < 4; ++j) {
    a[j]      = (_Float16)q0[j];
    a[j + 4]  = (_Float16)q1[j];
    a[j + 8]  = (_Float16)q2[j];
    a[j + 12] = (_Float16)q3[j];
  }
  return a;
}

__device__ __forceinline__ v8f wmma_f16(v16h a, v16h b, v8f c) {
  return __builtin_amdgcn_wmma_f32_16x16x32_f16(false, a, false, b,
                                                (short)0, c, false, false);
}

// ---------------------------------------------------------------------------
// C[M,64] = act( A1 @ W1 [+ A2 @ W2] [+ bias] ), one wave per 16-row tile
// ---------------------------------------------------------------------------
__global__ void __launch_bounds__(256) gemm64_kernel(
    const float* __restrict__ A1, const float* __restrict__ W1,
    const float* __restrict__ A2, const float* __restrict__ W2,
    const float* __restrict__ bias, float* __restrict__ C,
    int M, int has2, int relu) {
  const int lane = threadIdx.x & 31;
  const int wid  = (blockIdx.x * blockDim.x + threadIdx.x) >> 5;
  const int nwav = (gridDim.x * blockDim.x) >> 5;

  v16h B1[4][2], B2[4][2];
#pragma unroll
  for (int t = 0; t < 4; ++t)
#pragma unroll
    for (int s = 0; s < 2; ++s)
      B1[t][s] = load_B_frag_f32(W1, t, s, lane);
  if (has2) {
#pragma unroll
    for (int t = 0; t < 4; ++t)
#pragma unroll
      for (int s = 0; s < 2; ++s)
        B2[t][s] = load_B_frag_f32(W2, t, s, lane);
  }

  const int ntiles = M >> 4;
  const int colL   = lane & 15;
  const int rOff   = (lane >> 4) << 3;
  for (int tile = wid; tile < ntiles; tile += nwav) {
    const int row0 = tile << 4;
    v16h a0 = load_A_frag_f32<false>(A1, row0, 0, lane);
    v16h a1 = load_A_frag_f32<false>(A1, row0, 1, lane);
    v8f acc[4] = {};
#pragma unroll
    for (int t = 0; t < 4; ++t) {
      acc[t] = wmma_f16(a0, B1[t][0], acc[t]);
      acc[t] = wmma_f16(a1, B1[t][1], acc[t]);
    }
    if (has2) {
      v16h c0 = load_A_frag_f32<false>(A2, row0, 0, lane);
      v16h c1 = load_A_frag_f32<false>(A2, row0, 1, lane);
#pragma unroll
      for (int t = 0; t < 4; ++t) {
        acc[t] = wmma_f16(c0, B2[t][0], acc[t]);
        acc[t] = wmma_f16(c1, B2[t][1], acc[t]);
      }
    }
    const int rbase = row0 + rOff;
#pragma unroll
    for (int t = 0; t < 4; ++t) {
      const int col = t * 16 + colL;
      const float bv = bias ? bias[col] : 0.f;
#pragma unroll
      for (int v = 0; v < 8; ++v) {
        float x = acc[t][v] + bv;
        if (relu) x = fmaxf(x, 0.f);
        C[(size_t)(rbase + v) * D + col] = x;   // RT: intra/inter/hs/hd re-read from L2
      }
    }
  }
}

// ---------------------------------------------------------------------------
// Fused edge kernel: m = relu(hs[src] + hd[dst] + Ef @ We); agg[dst] += m
// hs already contains b_msg. One wave per 16-edge tile.
// Edge features are a single-use 410MB/graph stream -> non-temporal loads.
// ---------------------------------------------------------------------------
__global__ void __launch_bounds__(256) edge_msg_kernel(
    const float* __restrict__ Ef, const float* __restrict__ We,
    const float* __restrict__ hs, const float* __restrict__ hd,
    const int* __restrict__ src, const int* __restrict__ dst,
    float* __restrict__ agg, int E) {
  const int lane = threadIdx.x & 31;
  const int wid  = (blockIdx.x * blockDim.x + threadIdx.x) >> 5;
  const int nwav = (gridDim.x * blockDim.x) >> 5;

  v16h B[4][2];
#pragma unroll
  for (int t = 0; t < 4; ++t)
#pragma unroll
    for (int s = 0; s < 2; ++s)
      B[t][s] = load_B_frag_f32(We, t, s, lane);

  const int ntiles = E >> 4;
  const int colL   = lane & 15;
  const int rOff   = (lane >> 4) << 3;
  for (int tile = wid; tile < ntiles; tile += nwav) {
    const int e0 = tile << 4;
    v16h a0 = load_A_frag_f32<true>(Ef, e0, 0, lane);
    v16h a1 = load_A_frag_f32<true>(Ef, e0, 1, lane);
    v8f acc[4] = {};
#pragma unroll
    for (int t = 0; t < 4; ++t) {
      acc[t] = wmma_f16(a0, B[t][0], acc[t]);
      acc[t] = wmma_f16(a1, B[t][1], acc[t]);
    }
#pragma unroll
    for (int v = 0; v < 8; ++v) {
      const int eidx = e0 + rOff + v;      // uniform within each 16-lane group
      const int s = src[eidx];
      const int d = dst[eidx];
      const float* ps = hs + (size_t)s * D;
      const float* pd = hd + (size_t)d * D;
      float* pa = agg + (size_t)d * D;
#pragma unroll
      for (int t = 0; t < 4; ++t) {
        const int col = t * 16 + colL;
        float x = acc[t][v] + ps[col] + pd[col];
        x = fmaxf(x, 0.f);
        atomicAdd(pa + col, x);            // L2-resident f32 atomic scatter-sum
      }
    }
  }
}

// ---------------------------------------------------------------------------
// Zero-fill (b128 stores)
// ---------------------------------------------------------------------------
__global__ void __launch_bounds__(256) zero_kernel(float* __restrict__ p, int n4) {
  int i = blockIdx.x * blockDim.x + threadIdx.x;
  const int stride = gridDim.x * blockDim.x;
  v4f z = {};
  for (; i < n4; i += stride) ((v4f*)p)[i] = z;
}

// ---------------------------------------------------------------------------
// 2-way attention combine: one wave32 per node, lane = hidden unit (HA==32)
// ---------------------------------------------------------------------------
__global__ void __launch_bounds__(256) attn_kernel(
    const float* __restrict__ intra, const float* __restrict__ inter,
    const float* __restrict__ W1,   // [64][32] row-major
    const float* __restrict__ b1,   // [32]
    const float* __restrict__ W2,   // [32]
    float* __restrict__ out, int N) {
  const int lane = threadIdx.x & 31;
  const int wid  = (blockIdx.x * blockDim.x + threadIdx.x) >> 5;
  const int nwav = (gridDim.x * blockDim.x) >> 5;

  for (int n = wid; n < N; n += nwav) {
    const float* a = intra + (size_t)n * D;
    const float* b = inter + (size_t)n * D;
    float ha = b1[lane], hb = b1[lane];
#pragma unroll 8
    for (int d = 0; d < D; ++d) {
      const float w = W1[d * HA + lane];   // coalesced
      ha += a[d] * w;                      // a[d]/b[d] uniform -> broadcast load
      hb += b[d] * w;
    }
    ha = tanhf(ha);
    hb = tanhf(hb);
    const float w2 = W2[lane];
    float wa = ha * w2, wb = hb * w2;
#pragma unroll
    for (int off = 16; off; off >>= 1) {   // wave32 tree reduction
      wa += __shfl_xor(wa, off, 32);
      wb += __shfl_xor(wb, off, 32);
    }
    const float m  = fmaxf(wa, wb);
    const float ea = __expf(wa - m), eb = __expf(wb - m);
    const float inv = 1.f / (ea + eb);
    const float ba = ea * inv, bb = eb * inv;
    const int d0 = lane * 2;
    // final output: written once, never re-read -> non-temporal stores
    __builtin_nontemporal_store(ba * a[d0]     + bb * b[d0],
                                &out[(size_t)n * D + d0]);
    __builtin_nontemporal_store(ba * a[d0 + 1] + bb * b[d0 + 1],
                                &out[(size_t)n * D + d0 + 1]);
  }
}

// ---------------------------------------------------------------------------
// Launcher
// ---------------------------------------------------------------------------
extern "C" void kernel_launch(void* const* d_in, const int* in_sizes, int n_in,
                              void* d_out, int out_size, void* d_ws, size_t ws_size,
                              hipStream_t stream) {
  const float* h        = (const float*)d_in[0];
  const float* efeat[2] = { (const float*)d_in[1], (const float*)d_in[2] };
  const int*   srcs[2]  = { (const int*)d_in[3], (const int*)d_in[5] };
  const int*   dsts[2]  = { (const int*)d_in[4], (const int*)d_in[6] };
  const float* W_src  = (const float*)d_in[7];
  const float* W_dst  = (const float*)d_in[8];
  const float* W_edge = (const float*)d_in[9];
  const float* b_msg  = (const float*)d_in[10];
  const float* W_self = (const float*)d_in[11];
  const float* W_agg  = (const float*)d_in[12];
  const float* b_node = (const float*)d_in[13];
  const float* a1w    = (const float*)d_in[14];
  const float* a1b    = (const float*)d_in[15];
  const float* a2w    = (const float*)d_in[16];
  float* out = (float*)d_out;

  const int N = in_sizes[0] / D;   // 100000 (divisible by 16)
  const int E = in_sizes[1] / D;   // 1600000 (divisible by 16)

  // workspace: hs, hd, agg (reused across graphs), intra, inter  (5 * N * 64 f32)
  float* hs    = (float*)d_ws;
  float* hd    = hs    + (size_t)N * D;
  float* agg   = hd    + (size_t)N * D;
  float* intra = agg   + (size_t)N * D;
  float* inter = intra + (size_t)N * D;
  float* nodeOut[2] = { intra, inter };

  const int threads    = 256;                    // 8 waves / block
  const int gemmBlocks = ((N >> 4) + 7) / 8;     // one wave per 16-row tile
  const int edgeBlocks = ((E >> 4) + 7) / 8;     // one wave per 16-edge tile

  for (int g = 0; g < 2; ++g) {
    const float* Wsrc_g  = W_src  + (size_t)g * D * D;
    const float* Wdst_g  = W_dst  + (size_t)g * D * D;
    const float* Wedge_g = W_edge + (size_t)g * D * D;
    const float* Wself_g = W_self + (size_t)g * D * D;
    const float* Wagg_g  = W_agg  + (size_t)g * D * D;

    // hs = h @ W_src + b_msg   (bias folded here);  hd = h @ W_dst
    gemm64_kernel<<<gemmBlocks, threads, 0, stream>>>(
        h, Wsrc_g, nullptr, nullptr, b_msg + g * D, hs, N, 0, 0);
    gemm64_kernel<<<gemmBlocks, threads, 0, stream>>>(
        h, Wdst_g, nullptr, nullptr, nullptr, hd, N, 0, 0);
    zero_kernel<<<1024, threads, 0, stream>>>(agg, (N * D) >> 2);
    // fused edge message + scatter-sum
    edge_msg_kernel<<<edgeBlocks, threads, 0, stream>>>(
        efeat[g], Wedge_g, hs, hd, srcs[g], dsts[g], agg, E);
    // node update: relu(h @ W_self + agg @ W_agg + b_node)
    gemm64_kernel<<<gemmBlocks, threads, 0, stream>>>(
        h, Wself_g, agg, Wagg_g, b_node + g * D, nodeOut[g], N, 1, 1);
  }

  attn_kernel<<<(N + 7) / 8, threads, 0, stream>>>(
      intra, inter, a1w, a1b, a2w, out, N);
}
